// W8A16LinearLayer_14611478741503
// MI455X (gfx1250) — compile-verified, matrix operand order checked by reading, not run on hardware
//
#include <hip/hip_runtime.h>
#include <stdint.h>

// ---------------------------------------------------------------------------
// W8A16 linear:  out[T,N] = A[T,K](f32) x W[N,K](int8)^T * scales[N] + bias[N]
// T=8192, K=4096, N=11008.
//
// MI455X plan (compute-bound, AI ~1.4 KFLOP/B):
//   pass 1: convert A -> bf16 (67MB) and W -> bf16 (90MB) into d_ws.
//           157MB fits the 192MB L2, so the GEMM re-reads bf16 from L2.
//   pass 2: bf16 WMMA GEMM, LDS staging via global_load_async_to_lds_b128
//           (ASYNCcnt), double-buffered, BK=64 (2 WMMA K-steps per barrier).
// Fallback: fused convert-in-staging kernel when ws is too small.
// ---------------------------------------------------------------------------

typedef __attribute__((ext_vector_type(16))) __bf16 v16bf;
typedef __attribute__((ext_vector_type(8)))  float  v8f;

#define TOKENS 8192
#define KDIM   4096
#define NDIM   11008
#define BM     128
#define BN     128

// ---------------------------- conversion helpers ---------------------------

__device__ __forceinline__ unsigned bf16pk_rhu(float a, float b) {
    // round-half-up f32->bf16, packed pair (cheap: 2 adds + merge)
    unsigned ua = __builtin_bit_cast(unsigned, a) + 0x8000u;
    unsigned ub = __builtin_bit_cast(unsigned, b) + 0x8000u;
    return (ua >> 16) | (ub & 0xFFFF0000u);
}

__device__ __forceinline__ unsigned bf16pk_exact(float a, float b) {
    // truncation; exact for small integers (int8 range)
    unsigned ua = __builtin_bit_cast(unsigned, a);
    unsigned ub = __builtin_bit_cast(unsigned, b);
    return (ua >> 16) | (ub & 0xFFFF0000u);
}

// --------------------------- prepass kernels -------------------------------

// A[f32] -> bf16 : 8 elems/thread
__global__ __launch_bounds__(256)
void cvt_act_kernel(const float* __restrict__ in, unsigned short* __restrict__ outp) {
    const size_t i = ((size_t)blockIdx.x * 256 + threadIdx.x) * 8;
    const float4* p = reinterpret_cast<const float4*>(in + i);
    const float4 f0 = p[0];
    const float4 f1 = p[1];
    uint4 r;
    r.x = bf16pk_rhu(f0.x, f0.y);
    r.y = bf16pk_rhu(f0.z, f0.w);
    r.z = bf16pk_rhu(f1.x, f1.y);
    r.w = bf16pk_rhu(f1.z, f1.w);
    *reinterpret_cast<uint4*>(outp + i) = r;
}

// W[int8] -> bf16 : 16 elems/thread (one b128 load, two b128 stores)
__global__ __launch_bounds__(256)
void cvt_wgt_kernel(const signed char* __restrict__ in, unsigned short* __restrict__ outp) {
    const size_t i = ((size_t)blockIdx.x * 256 + threadIdx.x) * 16;
    const int4 wv = *reinterpret_cast<const int4*>(in + i);
    const int wr[4] = { wv.x, wv.y, wv.z, wv.w };
    unsigned r[8];
    #pragma unroll
    for (int v = 0; v < 4; ++v) {
        const float f0 = (float)(signed char)(wr[v]);
        const float f1 = (float)(signed char)(wr[v] >> 8);
        const float f2 = (float)(signed char)(wr[v] >> 16);
        const float f3 = (float)(signed char)(wr[v] >> 24);
        r[v * 2 + 0] = bf16pk_exact(f0, f1);
        r[v * 2 + 1] = bf16pk_exact(f2, f3);
    }
    uint4* o = reinterpret_cast<uint4*>(outp + i);
    o[0] = (uint4){ r[0], r[1], r[2], r[3] };
    o[1] = (uint4){ r[4], r[5], r[6], r[7] };
}

// ------------------------- async-LDS bf16 GEMM -----------------------------

__device__ __forceinline__ void async_b128(unsigned lds_addr,
                                           unsigned long long gbase,
                                           unsigned voff) {
    // GVS mode: mem = SADDR + VADDR[31:0]; VDST = LDS byte address.
    asm volatile("global_load_async_to_lds_b128 %0, %1, %2"
                 :
                 : "v"(lds_addr), "v"(voff), "s"(gbase)
                 : "memory");
}

__device__ __forceinline__ void wait_async0() {
    asm volatile("s_wait_asynccnt 0x0" ::: "memory");
}

#define BKA 64                     // K per staged tile (2 WMMA K-steps)
#define NKI (KDIM / BKA)           // 64 iterations

__global__ __launch_bounds__(256, 2)
void w8a16_wmma_async_kernel(const unsigned short* __restrict__ Abf, // [T,K] bf16
                             const unsigned short* __restrict__ Wbf, // [N,K] bf16
                             const float* __restrict__ scales,
                             const float* __restrict__ bias,
                             float* __restrict__ out) {
    __shared__ __align__(32) unsigned short lds_a[2][BM * BKA]; // 2 x 16 KB
    __shared__ __align__(32) unsigned short lds_b[2][BN * BKA]; // 2 x 16 KB

    const int tid   = threadIdx.x;
    const int lane  = tid & 31;
    const int wave  = tid >> 5;
    const int waveM = wave & 3;   // 4 M-bands of 32 rows
    const int waveN = wave >> 2;  // 2 N-bands of 64 cols

    const int rowA0 = blockIdx.y * BM;
    const int rowW0 = blockIdx.x * BN;

    const unsigned long long aBase =
        (unsigned long long)(uintptr_t)(Abf + (size_t)rowA0 * KDIM);
    const unsigned long long wBase =
        (unsigned long long)(uintptr_t)(Wbf + (size_t)rowW0 * KDIM);

    const unsigned ldsA0 = (unsigned)(uintptr_t)&lds_a[0][0];
    const unsigned ldsA1 = (unsigned)(uintptr_t)&lds_a[1][0];
    const unsigned ldsB0 = (unsigned)(uintptr_t)&lds_b[0][0];
    const unsigned ldsB1 = (unsigned)(uintptr_t)&lds_b[1][0];

    // Tile = 128 rows x 64 bf16 (128B/row) = 1024 16B chunks; 4 per thread.
    auto stageAsync = [&](int kt, int buf) {
        const unsigned k0b  = (unsigned)(kt * BKA * 2);     // byte offset in K
        const unsigned la   = buf ? ldsA1 : ldsA0;
        const unsigned lb   = buf ? ldsB1 : ldsB0;
        #pragma unroll
        for (int i = 0; i < 4; ++i) {
            const int c   = tid + 256 * i;                  // chunk 0..1023
            const int row = c >> 3;
            const int cir = c & 7;
            const unsigned goff = (unsigned)(row * (KDIM * 2)) + k0b + cir * 16;
            async_b128(la + (unsigned)(c * 16), aBase, goff);
            async_b128(lb + (unsigned)(c * 16), wBase, goff);
        }
    };

    v8f acc[2][4];
    #pragma unroll
    for (int mi = 0; mi < 2; ++mi)
        #pragma unroll
        for (int ni = 0; ni < 4; ++ni)
            acc[mi][ni] = (v8f){};

    stageAsync(0, 0);
    wait_async0();
    __syncthreads();

    for (int kt = 0; kt < NKI; ++kt) {
        const int cur = kt & 1;
        if (kt + 1 < NKI) stageAsync(kt + 1, cur ^ 1);

        #pragma unroll
        for (int ks = 0; ks < 2; ++ks) {                    // two K=32 steps
            const int kb = ks * 32 + ((lane >> 4) * 16);    // per-lane K half
            v16bf af[2], bfr[4];
            #pragma unroll
            for (int mi = 0; mi < 2; ++mi) {
                const int r = waveM * 32 + mi * 16 + (lane & 15);
                af[mi] = *reinterpret_cast<const v16bf*>(&lds_a[cur][r * BKA + kb]);
            }
            #pragma unroll
            for (int ni = 0; ni < 4; ++ni) {
                const int r = waveN * 64 + ni * 16 + (lane & 15);
                bfr[ni] = *reinterpret_cast<const v16bf*>(&lds_b[cur][r * BKA + kb]);
            }
            #pragma unroll
            for (int mi = 0; mi < 2; ++mi)
                #pragma unroll
                for (int ni = 0; ni < 4; ++ni)
                    acc[mi][ni] = __builtin_amdgcn_wmma_f32_16x16x32_bf16(
                        false, af[mi], false, bfr[ni],
                        (short)0, acc[mi][ni], false, false);
        }

        wait_async0();      // own async copies into `nxt` have landed in LDS
        __syncthreads();    // everyone's copies visible; done reading `cur`
    }

    // Epilogue: scale + bias, non-temporal stores (don't pollute L2).
    const int colBase = rowW0 + waveN * 64;
    #pragma unroll
    for (int ni = 0; ni < 4; ++ni) {
        const int col = colBase + ni * 16 + (lane & 15);
        const float s = scales[col];
        const float z = bias[col];
        #pragma unroll
        for (int mi = 0; mi < 2; ++mi) {
            const int rbase = rowA0 + waveM * 32 + mi * 16 + ((lane >> 4) << 3);
            #pragma unroll
            for (int r = 0; r < 8; ++r) {
                const float v = acc[mi][ni][r] * s + z;
                __builtin_nontemporal_store(v, out + (size_t)(rbase + r) * NDIM + col);
            }
        }
    }
}

// ------------------- fallback: fused conversion GEMM -----------------------

#define BKF 32
#define NKF (KDIM / BKF)

__device__ __forceinline__ unsigned short f32_to_bf16_rne(float f) {
    unsigned u = __builtin_bit_cast(unsigned, f);
    u += 0x7FFFu + ((u >> 16) & 1u);
    return (unsigned short)(u >> 16);
}

__global__ __launch_bounds__(256, 2)
void w8a16_wmma_fused_kernel(const float* __restrict__ A,
                             const signed char* __restrict__ W,
                             const float* __restrict__ scales,
                             const float* __restrict__ bias,
                             float* __restrict__ out) {
    __shared__ __align__(32) unsigned short lds_a[2][BM * BKF];
    __shared__ __align__(32) unsigned short lds_b[2][BN * BKF];

    const int tid   = threadIdx.x;
    const int lane  = tid & 31;
    const int wave  = tid >> 5;
    const int waveM = wave & 3;
    const int waveN = wave >> 2;

    const int rowA0 = blockIdx.y * BM;
    const int rowW0 = blockIdx.x * BN;

    const int sRow  = tid >> 1;
    const int sHalf = (tid & 1) * 16;

    const float*       aSrc = A + (size_t)(rowA0 + sRow) * KDIM + sHalf;
    const signed char* wSrc = W + (size_t)(rowW0 + sRow) * KDIM + sHalf;

    auto stage = [&](int kt, int buf) {
        const int k0 = kt * BKF;
        const float4* ap = reinterpret_cast<const float4*>(aSrc + k0);
        unsigned short* la = &lds_a[buf][sRow * BKF + sHalf];
        #pragma unroll
        for (int v = 0; v < 4; ++v) {
            float4 f = ap[v];
            la[v * 4 + 0] = f32_to_bf16_rne(f.x);
            la[v * 4 + 1] = f32_to_bf16_rne(f.y);
            la[v * 4 + 2] = f32_to_bf16_rne(f.z);
            la[v * 4 + 3] = f32_to_bf16_rne(f.w);
        }
        const int4 wv = *reinterpret_cast<const int4*>(wSrc + k0);
        const int wr[4] = { wv.x, wv.y, wv.z, wv.w };
        unsigned short* lb = &lds_b[buf][sRow * BKF + sHalf];
        #pragma unroll
        for (int v = 0; v < 4; ++v)
            #pragma unroll
            for (int b = 0; b < 4; ++b)
                lb[v * 4 + b] = (unsigned short)
                    (__builtin_bit_cast(unsigned,
                        (float)(signed char)(wr[v] >> (8 * b))) >> 16);
        if (k0 + 2 * BKF < KDIM) {
            __builtin_prefetch(aSrc + k0 + 2 * BKF, 0, 3);
            __builtin_prefetch(wSrc + k0 + 2 * BKF, 0, 3);
        }
    };

    v8f acc[2][4];
    #pragma unroll
    for (int mi = 0; mi < 2; ++mi)
        #pragma unroll
        for (int ni = 0; ni < 4; ++ni)
            acc[mi][ni] = (v8f){};

    stage(0, 0);
    __syncthreads();

    for (int kt = 0; kt < NKF; ++kt) {
        const int cur = kt & 1;
        if (kt + 1 < NKF) stage(kt + 1, cur ^ 1);

        const int kb = (lane >> 4) * 16;
        v16bf af[2], bfr[4];
        #pragma unroll
        for (int mi = 0; mi < 2; ++mi) {
            const int r = waveM * 32 + mi * 16 + (lane & 15);
            af[mi] = *reinterpret_cast<const v16bf*>(&lds_a[cur][r * BKF + kb]);
        }
        #pragma unroll
        for (int ni = 0; ni < 4; ++ni) {
            const int r = waveN * 64 + ni * 16 + (lane & 15);
            bfr[ni] = *reinterpret_cast<const v16bf*>(&lds_b[cur][r * BKF + kb]);
        }
        #pragma unroll
        for (int mi = 0; mi < 2; ++mi)
            #pragma unroll
            for (int ni = 0; ni < 4; ++ni)
                acc[mi][ni] = __builtin_amdgcn_wmma_f32_16x16x32_bf16(
                    false, af[mi], false, bfr[ni],
                    (short)0, acc[mi][ni], false, false);

        __syncthreads();
    }

    const int colBase = rowW0 + waveN * 64;
    #pragma unroll
    for (int ni = 0; ni < 4; ++ni) {
        const int col = colBase + ni * 16 + (lane & 15);
        const float s = scales[col];
        const float z = bias[col];
        #pragma unroll
        for (int mi = 0; mi < 2; ++mi) {
            const int rbase = rowA0 + waveM * 32 + mi * 16 + ((lane >> 4) << 3);
            #pragma unroll
            for (int r = 0; r < 8; ++r) {
                const float v = acc[mi][ni][r] * s + z;
                __builtin_nontemporal_store(v, out + (size_t)(rbase + r) * NDIM + col);
            }
        }
    }
}

// ------------------------------- launcher ----------------------------------

extern "C" void kernel_launch(void* const* d_in, const int* in_sizes, int n_in,
                              void* d_out, int out_size, void* d_ws, size_t ws_size,
                              hipStream_t stream) {
    const float*       input  = (const float*)d_in[0];
    const signed char* w8     = (const signed char*)d_in[1];
    const float*       scales = (const float*)d_in[2];
    const float*       bias   = (const float*)d_in[3];
    float*             out    = (float*)d_out;

    const size_t nA = (size_t)TOKENS * KDIM;            // 33,554,432
    const size_t nW = (size_t)NDIM  * KDIM;             // 45,088,768
    const size_t need = (nA + nW) * sizeof(unsigned short);

    dim3 grid(NDIM / BN, TOKENS / BM);                  // 86 x 64

    if (ws_size >= need) {
        unsigned short* aBf = (unsigned short*)d_ws;
        unsigned short* wBf = aBf + nA;
        cvt_act_kernel<<<(int)(nA / (256 * 8)),  256, 0, stream>>>(input, aBf);
        cvt_wgt_kernel<<<(int)(nW / (256 * 16)), 256, 0, stream>>>(w8, wBf);
        w8a16_wmma_async_kernel<<<grid, 256, 0, stream>>>(aBf, wBf, scales, bias, out);
    } else {
        w8a16_wmma_fused_kernel<<<grid, 256, 0, stream>>>(input, w8, scales, bias, out);
    }
}
